// Attention_23502061044282
// MI455X (gfx1250) — matrix operand order repaired and checked by reading, not compiled, not run on hardware
//
#include <hip/hip_runtime.h>
#include <hip/hip_bf16.h>
#include <stdint.h>

typedef __attribute__((ext_vector_type(8)))  _Float16 v8h;
typedef __attribute__((ext_vector_type(16))) _Float16 v16h;
typedef __attribute__((ext_vector_type(8)))  float    v8f;

#define BM 128
#define BN 128
#define BK 64   // two K=32 WMMA steps per k-tile; double buffered via async-to-LDS

// Low 32 bits of a generic pointer into LDS are the wave-relative LDS offset
// (ISA 10.2 aperture calc).
__device__ __forceinline__ uint32_t lds_addr32(const void* p) {
  return (uint32_t)(uintptr_t)p;
}

// CDNA5 async copy: global -> LDS, tracked with ASYNCcnt, no staging VGPRs.
__device__ __forceinline__ void async_b128_to_lds(uint32_t lds, const void* gptr) {
  asm volatile("global_load_async_to_lds_b128 %0, %1, off"
               :: "v"(lds), "v"((uint64_t)(uintptr_t)gptr)
               : "memory");
}

__device__ __forceinline__ void wait_async0() {
  asm volatile("s_wait_asynccnt 0" ::: "memory");
}

// ---------------------------------------------------------------------------
// Kernel 1: per-token asymmetric int8 fake-quant (torchao per_token_dynamic),
// output as f16. One 256-thread block per token row of K=2048 (8 elems/thread).
// ---------------------------------------------------------------------------
__global__ void __launch_bounds__(256)
quant_act_f16_kernel(const float* __restrict__ x, _Float16* __restrict__ a, int K) {
  const int row = blockIdx.x;
  const int tid = threadIdx.x;
  const float* xr = x + (size_t)row * K;
  _Float16* ar = a + (size_t)row * K;

  float4 v0 = ((const float4*)xr)[tid * 2 + 0];
  float4 v1 = ((const float4*)xr)[tid * 2 + 1];

  float mn = fminf(0.0f, fminf(fminf(v0.x, v0.y), fminf(v0.z, v0.w)));
  mn = fminf(mn, fminf(fminf(v1.x, v1.y), fminf(v1.z, v1.w)));
  float mx = fmaxf(0.0f, fmaxf(fmaxf(v0.x, v0.y), fmaxf(v0.z, v0.w)));
  mx = fmaxf(mx, fmaxf(fmaxf(v1.x, v1.y), fmaxf(v1.z, v1.w)));

  __shared__ float smin[256];
  __shared__ float smax[256];
  smin[tid] = mn; smax[tid] = mx;
  __syncthreads();
  #pragma unroll
  for (int s = 128; s > 0; s >>= 1) {
    if (tid < s) {
      smin[tid] = fminf(smin[tid], smin[tid + s]);
      smax[tid] = fmaxf(smax[tid], smax[tid + s]);
    }
    __syncthreads();
  }
  const float scale = fmaxf((smax[0] - smin[0]) * (1.0f / 255.0f), 1.1920929e-07f);
  const float inv   = 1.0f / scale;
  const float zp    = fminf(fmaxf(-128.0f - rintf(smin[0] * inv), -128.0f), 127.0f);

  float e[8] = {v0.x, v0.y, v0.z, v0.w, v1.x, v1.y, v1.z, v1.w};
  v8h o;
  #pragma unroll
  for (int j = 0; j < 8; ++j) {
    float q = fminf(fmaxf(rintf(e[j] * inv) + zp, -128.0f), 127.0f);
    o[j] = (_Float16)((q - zp) * scale);
  }
  *(v8h*)(ar + tid * 8) = o;
}

// ---------------------------------------------------------------------------
// Kernel 2: group-32 int4 dequant of W -> f16 panel (8 elems/thread; an
// aligned 8-chunk never crosses a group-of-32 boundary).
// ---------------------------------------------------------------------------
__global__ void __launch_bounds__(256)
dequant_w_f16_kernel(const int* __restrict__ w, const float* __restrict__ sc,
                     const float* __restrict__ zp, _Float16* __restrict__ out,
                     int K, int G) {
  const size_t t  = (size_t)blockIdx.x * blockDim.x + threadIdx.x;
  const size_t k0 = t * 8;
  const size_t r  = k0 / (size_t)K;
  const int    k  = (int)(k0 % (size_t)K);
  const int    g  = k >> 5;  // GROUPSIZE = 32
  const float  s  = sc[r * G + g];
  const float  z  = zp[r * G + g];

  int4 w0 = ((const int4*)(w + k0))[0];
  int4 w1 = ((const int4*)(w + k0))[1];
  int wi[8] = {w0.x, w0.y, w0.z, w0.w, w1.x, w1.y, w1.z, w1.w};
  v8h o;
  #pragma unroll
  for (int j = 0; j < 8; ++j) o[j] = (_Float16)(((float)wi[j] - z) * s);
  *(v8h*)(out + k0) = o;
}

// ---------------------------------------------------------------------------
// Kernel 3: C[M,N] = A[M,K] * W[N,K]^T via v_wmma_f32_16x16x32_f16.
// 128x128 block tile, BK=64, 4 wave32 waves (2x2), each wave owns a 64x64
// output tile (4x4 accumulators = 16 WMMAs per K=32 step, 1:1 wmma:ds ratio).
// Double-buffered LDS filled by async global->LDS copies; C staged through
// LDS for coalesced b128 stores. A/B buffers (64KB) union with C tile (64KB).
// ---------------------------------------------------------------------------
__global__ void __launch_bounds__(128)
gemm_f16_wmma_kernel(const _Float16* __restrict__ A, const _Float16* __restrict__ W,
                     float* __restrict__ C, int M, int N, int K) {
  __shared__ __align__(16) unsigned char smem[65536];
  _Float16* As0 = (_Float16*)smem;                 // BM*BK halves = 16KB
  _Float16* As1 = As0 + BM * BK;
  _Float16* Bs0 = As1 + BM * BK;
  _Float16* Bs1 = Bs0 + BN * BK;
  float* ctile  = (float*)smem;                    // 128*128 f32 = 64KB (epilogue)

  const int tid  = threadIdx.x;
  const int lane = tid & 31;
  const int wave = tid >> 5;      // 0..3
  const int wm   = wave >> 1;     // 0..1 : 64-row slab
  const int wn   = wave & 1;      // 0..1 : 64-col slab
  const int row0 = blockIdx.y * BM;
  const int col0 = blockIdx.x * BN;

  v8f acc[4][4];
  #pragma unroll
  for (int i = 0; i < 4; ++i)
    #pragma unroll
    for (int j = 0; j < 4; ++j) acc[i][j] = {};

  // 128 rows x 8 chunks(16B) = 1024 chunks per tile; 8 async loads per thread.
  auto issueTile = [&](const _Float16* __restrict__ src, int srcRow0,
                       _Float16* dst, int kt) {
    #pragma unroll
    for (int i = 0; i < 8; ++i) {
      int chunk = tid + i * 128;
      int r = chunk >> 3;
      int c = chunk & 7;
      const void* g = src + (size_t)(srcRow0 + r) * K + kt + c * 8;
      uint32_t l = lds_addr32(dst + r * BK + ((c ^ (r & 7)) * 8));
      async_b128_to_lds(l, g);
    }
  };

  // A fragment, ISA 16-bit A 16x32 layout: half-wave 0 -> K {0..7,16..23},
  // half-wave 1 -> K {8..15,24..31}; M = lane & 15.
  auto loadFragA = [&](const _Float16* s, int rowBase, int ko) -> v16h {
    int m = lane & 15, half = lane >> 4;
    int r = rowBase + m;
    int base = ko >> 3;                       // 0 or 4
    union { v16h f; v8h h[2]; } u;
    u.h[0] = *(const v8h*)(s + r * BK + (((base + half    ) ^ (r & 7)) * 8));
    u.h[1] = *(const v8h*)(s + r * BK + (((base + half + 2) ^ (r & 7)) * 8));
    return u.f;
  };

  // B fragment, ISA 16-bit B 32x16 layout: half-wave 0 -> K {0..7,16..23},
  // half-wave 1 -> K {8..15,24..31}; N = lane & 15 (row of W = column of B).
  auto loadFragB = [&](const _Float16* s, int colBase, int ko) -> v16h {
    int n = lane & 15, half = lane >> 4;
    int r = colBase + n;
    int base = (ko >> 3) + half;
    union { v16h f; v8h h[2]; } u;
    u.h[0] = *(const v8h*)(s + r * BK + (((base    ) ^ (r & 7)) * 8));
    u.h[1] = *(const v8h*)(s + r * BK + (((base + 2) ^ (r & 7)) * 8));
    return u.f;
  };

  issueTile(A, row0, As0, 0);
  issueTile(W, col0, Bs0, 0);
  wait_async0();
  __syncthreads();

  int buf = 0;
  for (int kt = 0; kt < K; kt += BK) {
    const _Float16* Acur = buf ? As1 : As0;
    const _Float16* Bcur = buf ? Bs1 : Bs0;
    if (kt + BK < K) {
      _Float16* Anx = buf ? As0 : As1;
      _Float16* Bnx = buf ? Bs0 : Bs1;
      issueTile(A, row0, Anx, kt + BK);
      issueTile(W, col0, Bnx, kt + BK);
    }
    #pragma unroll
    for (int inner = 0; inner < 2; ++inner) {
      const int ko = inner * 32;
      v16h a[4], b[4];
      #pragma unroll
      for (int i = 0; i < 4; ++i) a[i] = loadFragA(Acur, wm * 64 + i * 16, ko);
      #pragma unroll
      for (int j = 0; j < 4; ++j) b[j] = loadFragB(Bcur, wn * 64 + j * 16, ko);

      #pragma unroll
      for (int i = 0; i < 4; ++i)
        #pragma unroll
        for (int j = 0; j < 4; ++j)
          acc[i][j] = __builtin_amdgcn_wmma_f32_16x16x32_f16(
              false, a[i], false, b[j], (short)0, acc[i][j], false, false);
    }
    wait_async0();     // prefetch for next tile complete (per-wave)
    __syncthreads();   // all waves' async copies visible; frag reads done
    buf ^= 1;
  }

  // Epilogue: stage C tile in LDS (overwrites A/B buffers; safe after barrier),
  // then fully-coalesced b128 row stores.
  // C/D layout: lane holds column n = lane&15; VGPR v holds row v + 8*(lane>>4).
  #pragma unroll
  for (int i = 0; i < 4; ++i) {
    #pragma unroll
    for (int j = 0; j < 4; ++j) {
      int r0 = wm * 64 + i * 16 + (lane >> 4) * 8;
      int cc = wn * 64 + j * 16 + (lane & 15);
      #pragma unroll
      for (int v = 0; v < 8; ++v)
        ctile[(r0 + v) * BN + cc] = acc[i][j][v];
    }
  }
  __syncthreads();
  #pragma unroll
  for (int it = 0; it < 32; ++it) {
    int chunk = tid + it * 128;         // 128 rows x 32 float4 chunks
    int r = chunk >> 5;
    int c = chunk & 31;
    *(float4*)(C + (size_t)(row0 + r) * N + col0 + c * 4) =
        *(const float4*)(ctile + r * BN + c * 4);
  }
}

// ---------------------------------------------------------------------------
extern "C" void kernel_launch(void* const* d_in, const int* in_sizes, int n_in,
                              void* d_out, int out_size, void* d_ws, size_t ws_size,
                              hipStream_t stream) {
  const float* x   = (const float*)d_in[0];
  const int*   w   = (const int*)d_in[1];
  const float* wsc = (const float*)d_in[2];
  const float* wzp = (const float*)d_in[3];
  float* out = (float*)d_out;

  const int K = 2048, N = 2048;
  const int G = K / 32;
  const int M = in_sizes[0] / K;   // B*S = 8192

  _Float16* Ah = (_Float16*)d_ws;            // M*K halves  (~33.5 MB)
  _Float16* Wh = Ah + (size_t)M * K;         // N*K halves  (~8 MB)

  quant_act_f16_kernel<<<M, 256, 0, stream>>>(x, Ah, K);

  int wThreads = (N * K) / 8;
  dequant_w_f16_kernel<<<wThreads / 256, 256, 0, stream>>>(w, wsc, wzp, Wh, K, G);

  dim3 grid(N / BN, M / BM);
  gemm_f16_wmma_kernel<<<grid, 128, 0, stream>>>(Ah, Wh, out, M, N, K);
}
